// BiLSTM_CRF_38594576122299
// MI455X (gfx1250) — matrix operand order repaired and checked by reading, not compiled, not run on hardware
//
#include <hip/hip_runtime.h>
#include <hip/hip_bf16.h>

typedef __attribute__((ext_vector_type(16))) _Float16 v16h;
typedef __attribute__((ext_vector_type(8)))  _Float16 v8h;
typedef __attribute__((ext_vector_type(8)))  float    v8f;

#define SLEN  1024
#define BATCH 64
#define EMBD  256
#define HD    256      // per-direction hidden
#define HID2  512
#define NTAG  12
#define NEGV  (-10000.0f)

__device__ __forceinline__ float sigf(float x) { return 1.0f / (1.0f + __expf(-x)); }

// Build a 16x32 f16 A-fragment from a row-major LDS/global row.
// Layout assumption (ISA 7.12.2): lanes 0-15 -> M=lane, K 0..7 & 16..23;
// lanes 16-31 -> M=lane-16, K 8..15 & 24..31.
__device__ __forceinline__ v16h make_afrag(const _Float16* p) {
  v8h lo = *(const v8h*)p;        // K base
  v8h hi = *(const v8h*)(p + 16); // K base + 16
  v16h a;
#pragma unroll
  for (int i = 0; i < 8; ++i) { a[i] = lo[i]; a[8 + i] = hi[i]; }
  return a;
}

// Async global -> LDS copy of 16 bytes (CDNA5 ASYNCcnt path).
// VDST carries the LDS byte address (low 32 bits of the generic pointer).
__device__ __forceinline__ void async_copy_b128(_Float16* lds_dst,
                                                const _Float16* gsrc) {
  asm volatile("global_load_async_to_lds_b128 %0, %1, off"
               :: "v"((unsigned)(size_t)lds_dst),
                  "v"((unsigned long long)(size_t)gsrc)
               : "memory");
}
__device__ __forceinline__ void wait_async0() {
  asm volatile("s_wait_asynccnt 0x0" ::: "memory");
}

// ---------------------------------------------------------------------------
// 1) Embedding gather + f32 -> f16.  x0 layout: (s, b, 256), row = s*64+b.
// ---------------------------------------------------------------------------
__global__ void embed_kernel(const int* __restrict__ sent,
                             const float* __restrict__ emb,
                             _Float16* __restrict__ x0) {
  const int r = blockIdx.x * 8 + (threadIdx.x >> 5); // row index
  const int e = (threadIdx.x & 31) * 8;
  const int s = r >> 6, b = r & 63;
  const int tok = sent[b * SLEN + s];
  const float* src = emb + (size_t)tok * EMBD + e;
  v8h v;
#pragma unroll
  for (int i = 0; i < 8; ++i) v[i] = (_Float16)src[i];
  *(v8h*)&x0[(size_t)r * EMBD + e] = v;
}

// ---------------------------------------------------------------------------
// 2) Weight packing: fuse [Wih; Whh] (K x 1024) into WMMA B-fragment order.
//    half index = ((nt*KT + kt)*32 + lane)*16 + j
//    maps to element n = nt*16 + (lane&15), k = kt*32 + (lane>>4)*8 + (j&7) + (j>>3)*16.
// ---------------------------------------------------------------------------
__global__ void pack_w_kernel(const float* __restrict__ wih,
                              const float* __restrict__ whh,
                              _Float16* __restrict__ out,
                              int IN, int KT, int total) {
  const int idx = blockIdx.x * blockDim.x + threadIdx.x;
  if (idx >= total) return;
  const int j    = idx & 15;
  const int lane = (idx >> 4) & 31;
  const int tile = idx >> 9;
  const int kt = tile % KT;
  const int nt = tile / KT;
  const int n = nt * 16 + (lane & 15);
  const int k = kt * 32 + (lane >> 4) * 8 + (j & 7) + (j >> 3) * 16;
  const float v = (k < IN) ? wih[(size_t)n * IN + k]
                           : whh[(size_t)n * HD + (k - IN)];
  out[idx] = (_Float16)v;
}

__global__ void bias_kernel(const float* __restrict__ a,
                            const float* __restrict__ b,
                            float* __restrict__ o, int n) {
  const int i = blockIdx.x * blockDim.x + threadIdx.x;
  if (i < n) o[i] = a[i] + b[i];
}

// w_out (12 x 512) -> one 16-wide column strip of B fragments, zero padded.
__global__ void pack_wout_kernel(const float* __restrict__ w,
                                 _Float16* __restrict__ out) {
  const int idx = blockIdx.x * blockDim.x + threadIdx.x;
  if (idx >= 16 * 512) return;
  const int j    = idx & 15;
  const int lane = (idx >> 4) & 31;
  const int kt   = idx >> 9;
  const int n = lane & 15;
  const int k = kt * 32 + (lane >> 4) * 8 + (j & 7) + (j >> 3) * 16;
  out[idx] = (_Float16)((n < NTAG) ? w[(size_t)n * HID2 + k] : 0.0f);
}

// ---------------------------------------------------------------------------
// 3) Persistent bidirectional LSTM step kernel. grid.x = 2 (fwd, bwd).
//    One 32-wave workgroup owns the whole M=64 x N=1024 step GEMM.
//    Wave w owns N-tiles {g*16 + (w&15), g=0..3} and M-tiles {2*(w>>4)+..}:
//    all four gates (i,f,g,o) of its unit set stay in-register, so the cell
//    state c is register-resident across all 1024 steps (no gate LDS buffer).
//    x_t is double-buffered in LDS and staged with async global->LDS loads.
// ---------------------------------------------------------------------------
template <int IN>
__global__ void __launch_bounds__(1024, 1)
lstm_dir_kernel(const _Float16* __restrict__ x,      // (S, 64, IN)
                const _Float16* __restrict__ wpack,  // 2 dirs, packed
                const float* __restrict__ biasg,     // 2 dirs x 1024
                _Float16* __restrict__ y)            // (S, 64, 512)
{
  constexpr int K   = IN + HD;
  constexpr int KT  = K / 32;
  constexpr int KTX = IN / 32;   // x-part k-tiles
  constexpr int KTH = HD / 32;   // h-part k-tiles (8)
  constexpr int XCH = BATCH * IN / 8; // 16B chunks per x slab

  const int dir = blockIdx.x;
  const bool rev = (dir != 0);
  const _Float16* wp = wpack + (size_t)dir * ((size_t)64 * KT * 512);
  const float* bfused = biasg + dir * 1024;

  extern __shared__ __align__(16) char smem[];
  _Float16* xbuf[2];
  xbuf[0] = (_Float16*)smem;              // 64*IN halfs
  xbuf[1] = xbuf[0] + BATCH * IN;         // 64*IN halfs
  _Float16* hbuf = xbuf[1] + BATCH * IN;  // 64*256 halfs

  const int tid    = threadIdx.x;
  const int lane   = tid & 31;
  const int wid    = tid >> 5;
  const int ngroup = wid & 15;        // unit-column group: j in [16*ngroup, +16)
  const int mhalf  = wid >> 4;        // M-tile pair {2*mhalf, 2*mhalf+1}
  const int m0     = lane & 15;
  const int kb     = (lane >> 4) * 8;
  const int mhi    = (lane >> 4) * 8;
  const int j      = ngroup * 16 + (lane & 15);

  // Per-lane fused biases for this lane's unit column (gate order i,f,g,o).
  const float b_i = bfused[j];
  const float b_f = bfused[256 + j];
  const float b_g = bfused[512 + j];
  const float b_o = bfused[768 + j];

  // Register-resident cell state for 2 M-tiles x 8 rows.
  v8f cst[2];
  const v8f vzero = {};
  cst[0] = vzero; cst[1] = vzero;

  // Zero h, then async-stage x for step 0 into xbuf[0].
  for (int u = tid; u < BATCH * HD; u += 1024) hbuf[u] = (_Float16)0.0f;
  {
    const int s0 = rev ? (SLEN - 1) : 0;
    const _Float16* xs = x + (size_t)s0 * BATCH * IN;
    for (int idx = tid; idx < XCH; idx += 1024)
      async_copy_b128(xbuf[0] + idx * 8, xs + idx * 8);
  }
  __syncthreads();

  for (int step = 0; step < SLEN; ++step) {
    const int s = rev ? (SLEN - 1 - step) : step;
    const int cur = step & 1;
    const _Float16* xcur = xbuf[cur];

    // Wait for this step's async x slab (own wave), then rendezvous.
    wait_async0();
    __syncthreads();

    // Kick off next step's x slab into the other buffer.
    if (step + 1 < SLEN) {
      const int sn = rev ? (SLEN - 2 - step) : (step + 1);
      const _Float16* xs = x + (size_t)sn * BATCH * IN;
      _Float16* dst = xbuf[cur ^ 1];
      for (int idx = tid; idx < XCH; idx += 1024)
        async_copy_b128(dst + idx * 8, xs + idx * 8);
    }

    // --- step GEMM: gates = [x_t; h] @ [Wih; Whh]  (M=64, N=1024, K) ---
    v8f acc[2][4];
#pragma unroll
    for (int mt2 = 0; mt2 < 2; ++mt2)
#pragma unroll
      for (int g = 0; g < 4; ++g) acc[mt2][g] = vzero;

    // x-part of K
    for (int kt = 0; kt < KTX; ++kt) {
      v16h afrag[2];
#pragma unroll
      for (int mt2 = 0; mt2 < 2; ++mt2) {
        const int mt = 2 * mhalf + mt2;
        afrag[mt2] = make_afrag(&xcur[(mt * 16 + m0) * IN + kt * 32 + kb]);
      }
#pragma unroll
      for (int g = 0; g < 4; ++g) {
        const int nt = g * 16 + ngroup;
        const v16h* bp = (const v16h*)wp + ((size_t)(nt * KT + kt) * 32 + lane);
        __builtin_prefetch((const void*)(bp + 32), 0, 1);  // next kt slice
        v16h bfrag = *bp;
#pragma unroll
        for (int mt2 = 0; mt2 < 2; ++mt2)
          acc[mt2][g] = __builtin_amdgcn_wmma_f32_16x16x32_f16(
              false, afrag[mt2], false, bfrag, (short)0, acc[mt2][g],
              false, false);
      }
    }
    // h-part of K
    for (int kh = 0; kh < KTH; ++kh) {
      v16h afrag[2];
#pragma unroll
      for (int mt2 = 0; mt2 < 2; ++mt2) {
        const int mt = 2 * mhalf + mt2;
        afrag[mt2] = make_afrag(&hbuf[(mt * 16 + m0) * HD + kh * 32 + kb]);
      }
#pragma unroll
      for (int g = 0; g < 4; ++g) {
        const int nt = g * 16 + ngroup;
        const v16h* bp =
            (const v16h*)wp + ((size_t)(nt * KT + KTX + kh) * 32 + lane);
        __builtin_prefetch((const void*)(bp + 32), 0, 1);
        v16h bfrag = *bp;
#pragma unroll
        for (int mt2 = 0; mt2 < 2; ++mt2)
          acc[mt2][g] = __builtin_amdgcn_wmma_f32_16x16x32_f16(
              false, afrag[mt2], false, bfrag, (short)0, acc[mt2][g],
              false, false);
      }
    }

    // --- in-register LSTM cell update (gate order i, f, g, o) ---
    _Float16* yrow = y + (size_t)s * BATCH * HID2 + dir * HD;
#pragma unroll
    for (int mt2 = 0; mt2 < 2; ++mt2) {
      const int mt = 2 * mhalf + mt2;
#pragma unroll
      for (int r = 0; r < 8; ++r) {
        const int b = mt * 16 + mhi + r;
        const float gi = acc[mt2][0][r] + b_i;
        const float gf = acc[mt2][1][r] + b_f;
        const float gg = acc[mt2][2][r] + b_g;
        const float go = acc[mt2][3][r] + b_o;
        const float c = sigf(gf) * cst[mt2][r] + sigf(gi) * tanhf(gg);
        cst[mt2][r] = c;
        const _Float16 hh = (_Float16)(sigf(go) * tanhf(c));
        hbuf[b * HD + j] = hh;             // recurrent input for next step
        yrow[(size_t)b * HID2 + j] = hh;   // layer output (concat layout)
      }
    }
    __syncthreads();  // h complete before next step's GEMM reads it
  }
}

// ---------------------------------------------------------------------------
// 4) Projection: feats(b,s,16) = Y(s,b,512) @ w_out^T + b_out, N padded to 16.
//    One WMMA M-tile (16 rows, same s) per wave.
// ---------------------------------------------------------------------------
__global__ void __launch_bounds__(256)
proj_kernel(const _Float16* __restrict__ y2,
            const _Float16* __restrict__ wop,
            const float* __restrict__ b_out,
            float* __restrict__ feats) {
  const int lane = threadIdx.x & 31;
  const int wid  = threadIdx.x >> 5;
  const int mt = blockIdx.x * 8 + wid;     // 0..4095
  const int r0 = mt * 16;
  const int m0 = lane & 15;
  const int kb = (lane >> 4) * 8;
  v8f acc = {};
  for (int kt = 0; kt < 16; ++kt) {
    v16h a = make_afrag(y2 + (size_t)(r0 + m0) * HID2 + kt * 32 + kb);
    v16h b = *((const v16h*)wop + (size_t)kt * 32 + lane);
    acc = __builtin_amdgcn_wmma_f32_16x16x32_f16(
        false, a, false, b, (short)0, acc, false, false);
  }
  const int t = lane & 15;
  const float bo = (t < NTAG) ? b_out[t] : NEGV;
#pragma unroll
  for (int r = 0; r < 8; ++r) {
    const int row = r0 + r + ((lane >> 4) ? 8 : 0);
    const int s = row >> 6, b = row & 63;
    feats[((size_t)b * SLEN + s) * 16 + t] = acc[r] + bo;
  }
}

// ---------------------------------------------------------------------------
// 5) Viterbi: one block (1 wave) per batch element; backpointers kept in LDS.
// ---------------------------------------------------------------------------
__global__ void __launch_bounds__(32)
viterbi_kernel(const float* __restrict__ feats,   // (b, s, 16)
               const float* __restrict__ trans,   // (12, 12)
               int* __restrict__ out) {           // (b, s)
  __shared__ float fv[16];
  __shared__ float tr[NTAG * NTAG];
  __shared__ unsigned char bp[SLEN * 16];
  const int b = blockIdx.x;
  const int lane = threadIdx.x;

  for (int i = lane; i < NTAG * NTAG; i += 32) tr[i] = trans[i];
  if (lane < 16) fv[lane] = (lane == NTAG - 2) ? 0.0f : NEGV; // START
  __syncthreads();

  const float* fb = feats + (size_t)b * SLEN * 16;
  for (int s = 0; s < SLEN; ++s) {
    float best = -3.0e38f;
    int arg = 0;
    if (lane < NTAG) {
#pragma unroll
      for (int p = 0; p < NTAG; ++p) {
        const float sc = fv[p] + tr[lane * NTAG + p];
        if (sc > best) { best = sc; arg = p; }
      }
      best += fb[s * 16 + lane];
      bp[s * 16 + lane] = (unsigned char)arg;
    }
    __syncthreads();
    if (lane < NTAG) fv[lane] = best;
    __syncthreads();
  }

  if (lane == 0) {
    int cur = 0;
    float bb = -3.0e38f;
#pragma unroll
    for (int t = 0; t < NTAG; ++t) {
      const float sc = fv[t] + tr[(NTAG - 1) * NTAG + t]; // STOP row
      if (sc > bb) { bb = sc; cur = t; }
    }
    for (int s = SLEN - 1; s >= 0; --s) {
      out[(size_t)b * SLEN + s] = cur;
      cur = bp[s * 16 + cur];
    }
  }
}

// ---------------------------------------------------------------------------
// Host launcher
// ---------------------------------------------------------------------------
extern "C" void kernel_launch(void* const* d_in, const int* in_sizes, int n_in,
                              void* d_out, int out_size, void* d_ws, size_t ws_size,
                              hipStream_t stream) {
  (void)in_sizes; (void)n_in; (void)out_size; (void)ws_size;
  // Input order: sentences, lengths, tags, emb,
  //   l0f{wih,whh,bih,bhh}, l0b{...}, l1f{...}, l1b{...}, w_out, b_out, trans
  const int*   sent  = (const int*)d_in[0];
  const float* emb   = (const float*)d_in[3];
  const float* l0f_wih = (const float*)d_in[4];
  const float* l0f_whh = (const float*)d_in[5];
  const float* l0f_bih = (const float*)d_in[6];
  const float* l0f_bhh = (const float*)d_in[7];
  const float* l0b_wih = (const float*)d_in[8];
  const float* l0b_whh = (const float*)d_in[9];
  const float* l0b_bih = (const float*)d_in[10];
  const float* l0b_bhh = (const float*)d_in[11];
  const float* l1f_wih = (const float*)d_in[12];
  const float* l1f_whh = (const float*)d_in[13];
  const float* l1f_bih = (const float*)d_in[14];
  const float* l1f_bhh = (const float*)d_in[15];
  const float* l1b_wih = (const float*)d_in[16];
  const float* l1b_whh = (const float*)d_in[17];
  const float* l1b_bih = (const float*)d_in[18];
  const float* l1b_bhh = (const float*)d_in[19];
  const float* w_out = (const float*)d_in[20];
  const float* b_out = (const float*)d_in[21];
  const float* trans = (const float*)d_in[22];
  int* out = (int*)d_out;

  char* ws = (char*)d_ws;
  size_t off = 0;
  auto carve = [&](size_t bytes) -> void* {
    off = (off + 255) & ~(size_t)255;
    void* p = ws + off;
    off += bytes;
    return p;
  };

  const int KT0 = (EMBD + HD) / 32;   // 16
  const int KT1 = (HID2 + HD) / 32;   // 24
  _Float16* x0    = (_Float16*)carve((size_t)SLEN * BATCH * EMBD * 2);
  _Float16* y1    = (_Float16*)carve((size_t)SLEN * BATCH * HID2 * 2);
  _Float16* y2    = (_Float16*)carve((size_t)SLEN * BATCH * HID2 * 2);
  float*    feats = (float*)   carve((size_t)BATCH * SLEN * 16 * 4);
  _Float16* wp0   = (_Float16*)carve((size_t)2 * 64 * KT0 * 512 * 2);
  _Float16* wp1   = (_Float16*)carve((size_t)2 * 64 * KT1 * 512 * 2);
  _Float16* wpo   = (_Float16*)carve((size_t)16 * 512 * 2);
  float*    bias0 = (float*)   carve((size_t)2 * 1024 * 4);
  float*    bias1 = (float*)   carve((size_t)2 * 1024 * 4);

  // 1) embedding
  embed_kernel<<<SLEN * BATCH / 8, 256, 0, stream>>>(sent, emb, x0);

  // 2) weight packing
  const int tot0 = 64 * KT0 * 512;
  const int tot1 = 64 * KT1 * 512;
  pack_w_kernel<<<(tot0 + 255) / 256, 256, 0, stream>>>(l0f_wih, l0f_whh, wp0,          EMBD, KT0, tot0);
  pack_w_kernel<<<(tot0 + 255) / 256, 256, 0, stream>>>(l0b_wih, l0b_whh, wp0 + tot0,   EMBD, KT0, tot0);
  pack_w_kernel<<<(tot1 + 255) / 256, 256, 0, stream>>>(l1f_wih, l1f_whh, wp1,          HID2, KT1, tot1);
  pack_w_kernel<<<(tot1 + 255) / 256, 256, 0, stream>>>(l1b_wih, l1b_whh, wp1 + tot1,   HID2, KT1, tot1);
  bias_kernel<<<4, 256, 0, stream>>>(l0f_bih, l0f_bhh, bias0,        1024);
  bias_kernel<<<4, 256, 0, stream>>>(l0b_bih, l0b_bhh, bias0 + 1024, 1024);
  bias_kernel<<<4, 256, 0, stream>>>(l1f_bih, l1f_bhh, bias1,        1024);
  bias_kernel<<<4, 256, 0, stream>>>(l1b_bih, l1b_bhh, bias1 + 1024, 1024);
  pack_wout_kernel<<<32, 256, 0, stream>>>(w_out, wpo);

  // 3) recurrent layers (fwd + bwd concurrently via grid.x = 2)
  const size_t smem0 = ((size_t)2 * BATCH * EMBD + BATCH * HD) * 2;  // 96 KB
  const size_t smem1 = ((size_t)2 * BATCH * HID2 + BATCH * HD) * 2;  // 160 KB
  lstm_dir_kernel<EMBD><<<2, 1024, smem0, stream>>>(x0, wp0, bias0, y1);
  lstm_dir_kernel<HID2><<<2, 1024, smem1, stream>>>(y1, wp1, bias1, y2);

  // 4) projection to tag scores
  proj_kernel<<<512, 256, 0, stream>>>(y2, wpo, b_out, feats);

  // 5) Viterbi decode
  viterbi_kernel<<<BATCH, 32, 0, stream>>>(feats, trans, out);
}
